// LongformerAttentionMethod_44822278701217
// MI455X (gfx1250) — matrix operand order repaired and checked by reading, not compiled, not run on hardware
//
#include <hip/hip_runtime.h>
#include <hip/hip_bf16.h>
#include <math.h>

#define DEV __device__ __forceinline__

typedef __attribute__((ext_vector_type(16))) __bf16 v16bf;
typedef __attribute__((ext_vector_type(8)))  float  v8f;
typedef __attribute__((ext_vector_type(4)))  float  f4;

namespace {

constexpr int Bc = 2, Hc = 12, Sc = 2048, Dc = 64;

DEV v8f wmma_bf16(v16bf a, v16bf b, v8f c) {
  // D(16x16 f32) = A(16x32 bf16) * B(32x16 bf16) + C
  return __builtin_amdgcn_wmma_f32_16x16x32_bf16(false, a, false, b, (short)0, c,
                                                 false, false);
}

// 16-bit A-matrix 16x32 slot->K map (contiguous, matches B): lanes 0-15 hold K {0..15},
// lanes 16-31 hold K {16..31}; elem e -> vgpr e>>1, half e&1.
DEV int kA(int e, int g) { return e + 16 * g; }

// Load A operand: one q row per lane (m = lane&15), dims dbase + kA(e,g)
DEV v16bf loadA_f32(const float* __restrict__ row, int dbase, int g) {
  v16bf a;
#pragma unroll
  for (int e = 0; e < 16; ++e) a[e] = (__bf16)row[dbase + kA(e, g)];
  return a;
}

// B 32x16 in this code's contiguous k order (same as kA): lane = column N, elems e -> K = e + 16*(lane>>4).
// For scores (Q*K^T) the "K dim" is D, so this is a contiguous 16-float run.
DEV v16bf loadB_f32(const float* __restrict__ row, int dbase) {
  v16bf b;
#pragma unroll
  for (int e = 0; e < 16; ++e) b[e] = (__bf16)row[dbase + e];
  return b;
}

// Reductions across a 16-lane half (rows of the C/D tile live across 16 lanes)
DEV float hmax16(float x) {
  x = fmaxf(x, __shfl_xor(x, 1, 32));
  x = fmaxf(x, __shfl_xor(x, 2, 32));
  x = fmaxf(x, __shfl_xor(x, 4, 32));
  x = fmaxf(x, __shfl_xor(x, 8, 32));
  return x;
}
DEV float hsum16(float x) {
  x += __shfl_xor(x, 1, 32);
  x += __shfl_xor(x, 2, 32);
  x += __shfl_xor(x, 4, 32);
  x += __shfl_xor(x, 8, 32);
  return x;
}

} // namespace

// ---------------------------------------------------------------------------
// Kernel A: stable stream-compaction of the attention mask (one wave / batch)
// ---------------------------------------------------------------------------
__global__ __launch_bounds__(32) void compact_globals(const int* __restrict__ mask,
                                                      int* __restrict__ counts,
                                                      int* __restrict__ gidx) {
  const int b = blockIdx.x;
  const int lane = threadIdx.x;
  int base = 0;
  for (int i = 0; i < Sc; i += 32) {
    const int pred = mask[b * Sc + i + lane] > 0;
    const unsigned bm = (unsigned)__ballot(pred);   // wave32: low 32 bits
    const int ofs = __popc(bm & ((1u << lane) - 1u));
    if (pred) gidx[b * Sc + base + ofs] = i + lane;
    base += __popc(bm);
  }
  if (lane == 0) counts[b] = base;
}

// ---------------------------------------------------------------------------
// Kernel B: every query attends to the gathered global keys.
// One wave = one 16-row q tile. Two-pass softmax over 32-col chunks,
// writes normalized probs (incl. zero tail) and context.
// ---------------------------------------------------------------------------
__global__ __launch_bounds__(256) void longformer_global_attn(
    const float* __restrict__ q, const float* __restrict__ k,
    const float* __restrict__ v, const int* __restrict__ counts,
    const int* __restrict__ gidx, float* __restrict__ ctx,
    float* __restrict__ probs) {
  __shared__ __bf16 pLDS[8][16][40];  // per-wave 16x32 P tile (padded)

  const int tid = threadIdx.x;
  const int wave = tid >> 5, lane = tid & 31;
  const int g = lane >> 4, ln = lane & 15;

  const int qBlocks = Sc / 128;
  const int qblk = blockIdx.x % qBlocks;
  const int bh = blockIdx.x / qBlocks;
  const int b = bh / Hc;
  const int q0 = qblk * 128 + wave * 16;

  int count = counts[b];
  if (count < 1) count = 1;
  const int* gl = gidx + b * Sc;

  const float* qbase = q + (size_t)bh * Sc * Dc;
  const float* kbase = k + (size_t)bh * Sc * Dc;
  const float* vbase = v + (size_t)bh * Sc * Dc;

  const float* qrow = qbase + (size_t)(q0 + ln) * Dc;
  const v16bf aQ0 = loadA_f32(qrow, 0, g);
  const v16bf aQ1 = loadA_f32(qrow, 32, g);

  const float scale = 0.125f;  // 1/sqrt(64)
  float m8[8], l8[8];
#pragma unroll
  for (int r = 0; r < 8; ++r) { m8[r] = -__builtin_inff(); l8[r] = 0.f; }

  const int nChunks = (count + 31) / 32;

  // ---- pass 1: row max + softmax denominator ----
  for (int c = 0; c < nChunks; ++c) {
    const int gc = c * 32;
    v8f s[2];
#pragma unroll
    for (int t = 0; t < 2; ++t) {
      const int j = gc + t * 16 + ln;
      const int jc = (j < count) ? j : (count - 1);
      const float* krow = kbase + (size_t)gl[jc] * Dc;
      const v16bf b0 = loadB_f32(krow, 16 * g);
      const v16bf b1 = loadB_f32(krow, 32 + 16 * g);
      v8f sc = {};
      sc = wmma_bf16(aQ0, b0, sc);
      sc = wmma_bf16(aQ1, b1, sc);
      const bool valid = j < count;
#pragma unroll
      for (int r = 0; r < 8; ++r)
        s[t][r] = valid ? sc[r] * scale : -__builtin_inff();
    }
#pragma unroll
    for (int r = 0; r < 8; ++r) {
      const float cm = hmax16(fmaxf(s[0][r], s[1][r]));
      const float mn = fmaxf(m8[r], cm);
      const float f = __expf(m8[r] - mn);
      const float es = __expf(s[0][r] - mn) + __expf(s[1][r] - mn);
      l8[r] = l8[r] * f + hsum16(es);
      m8[r] = mn;
    }
  }

  // ---- pass 2: normalized probs out + context accumulation (P*V via WMMA) ----
  v8f acc[4] = {};
  float* probrow = probs + ((size_t)bh * Sc + q0) * (size_t)Sc;

  for (int c = 0; c < nChunks; ++c) {
    const int gc = c * 32;
    v8f p[2];
#pragma unroll
    for (int t = 0; t < 2; ++t) {
      const int j = gc + t * 16 + ln;
      const int jc = (j < count) ? j : (count - 1);
      const float* krow = kbase + (size_t)gl[jc] * Dc;
      const v16bf b0 = loadB_f32(krow, 16 * g);
      const v16bf b1 = loadB_f32(krow, 32 + 16 * g);
      v8f sc = {};
      sc = wmma_bf16(aQ0, b0, sc);
      sc = wmma_bf16(aQ1, b1, sc);
      const bool valid = j < count;
#pragma unroll
      for (int r = 0; r < 8; ++r) {
        const float sv = valid ? sc[r] * scale : -__builtin_inff();
        const float pv = __expf(sv - m8[r]) / l8[r];
        p[t][r] = pv;
        probrow[(size_t)(r + 8 * g) * Sc + (gc + t * 16 + ln)] = pv;
      }
    }
    // stage P (C/D layout) -> LDS -> A layout
#pragma unroll
    for (int t = 0; t < 2; ++t)
#pragma unroll
      for (int r = 0; r < 8; ++r)
        pLDS[wave][r + 8 * g][t * 16 + ln] = (__bf16)p[t][r];
    asm volatile("s_wait_dscnt 0" ::: "memory");
    v16bf aP;
#pragma unroll
    for (int e = 0; e < 16; ++e) aP[e] = pLDS[wave][ln][kA(e, g)];

    int vr[16];
#pragma unroll
    for (int e = 0; e < 16; ++e) {
      const int j = gc + e + 16 * g;  // B-operand K map: k = e + 16*g
      vr[e] = gl[(j < count) ? j : (count - 1)];
    }
#pragma unroll
    for (int nt = 0; nt < 4; ++nt) {
      v16bf bV;
#pragma unroll
      for (int e = 0; e < 16; ++e)
        bV[e] = (__bf16)vbase[(size_t)vr[e] * Dc + nt * 16 + ln];
      acc[nt] = wmma_bf16(aP, bV, acc[nt]);
    }
  }

  // ---- zero tail of probs rows (bulk of the bandwidth) ----
  const int z0 = nChunks * 32;
  if (z0 < Sc) {
    const f4 zz = {0.f, 0.f, 0.f, 0.f};
    for (int col = z0 + lane * 4; col < Sc; col += 128) {
#pragma unroll
      for (int r = 0; r < 16; ++r)
        __builtin_nontemporal_store(zz, (f4*)&probrow[(size_t)r * Sc + col]);
    }
  }

  // ---- context out ----
#pragma unroll
  for (int nt = 0; nt < 4; ++nt)
#pragma unroll
    for (int r = 0; r < 8; ++r)
      ctx[((size_t)bh * Sc + q0 + r + 8 * g) * Dc + nt * 16 + ln] = acc[nt][r];
}

// ---------------------------------------------------------------------------
// Kernel C: flash-style full attention for the global rows only; overwrites
// their context entries (probs untouched, matching the reference).
// ---------------------------------------------------------------------------
__global__ __launch_bounds__(256) void longformer_full_rows(
    const float* __restrict__ q, const float* __restrict__ k,
    const float* __restrict__ v, const int* __restrict__ counts,
    const int* __restrict__ gidx, float* __restrict__ ctx) {
  __shared__ __bf16 pLDS[8][16][40];

  const int tid = threadIdx.x;
  const int wave = tid >> 5, lane = tid & 31;
  const int g = lane >> 4, ln = lane & 15;

  const int blkPer = Sc / 128;
  const int tblk = blockIdx.x % blkPer;
  const int bh = blockIdx.x / blkPer;
  const int b = bh / Hc;
  const int count = counts[b];
  const int rowBase = tblk * 128 + wave * 16;
  if (rowBase >= count) return;  // no block barriers -> safe per-wave exit

  const int* gl = gidx + b * Sc;
  const float* qbase = q + (size_t)bh * Sc * Dc;
  const float* kbase = k + (size_t)bh * Sc * Dc;
  const float* vbase = v + (size_t)bh * Sc * Dc;

  const int qj = rowBase + ln;
  const float* qrow = qbase + (size_t)gl[(qj < count) ? qj : (count - 1)] * Dc;
  const v16bf aQ0 = loadA_f32(qrow, 0, g);
  const v16bf aQ1 = loadA_f32(qrow, 32, g);

  const float scale = 0.125f;
  float m8[8], l8[8];
#pragma unroll
  for (int r = 0; r < 8; ++r) { m8[r] = -__builtin_inff(); l8[r] = 0.f; }
  v8f acc[4] = {};

  for (int kc = 0; kc < Sc; kc += 32) {
    v8f s[2];
#pragma unroll
    for (int t = 0; t < 2; ++t) {
      const float* krow = kbase + (size_t)(kc + t * 16 + ln) * Dc;
      const v16bf b0 = loadB_f32(krow, 16 * g);
      const v16bf b1 = loadB_f32(krow, 32 + 16 * g);
      v8f sc = {};
      sc = wmma_bf16(aQ0, b0, sc);
      sc = wmma_bf16(aQ1, b1, sc);
#pragma unroll
      for (int r = 0; r < 8; ++r) s[t][r] = sc[r] * scale;
    }
#pragma unroll
    for (int r = 0; r < 8; ++r) {
      const float cm = hmax16(fmaxf(s[0][r], s[1][r]));
      const float mn = fmaxf(m8[r], cm);
      const float f = __expf(m8[r] - mn);
      const float p0 = __expf(s[0][r] - mn);
      const float p1 = __expf(s[1][r] - mn);
      l8[r] = l8[r] * f + hsum16(p0 + p1);
      m8[r] = mn;
#pragma unroll
      for (int nt = 0; nt < 4; ++nt) acc[nt][r] *= f;
      pLDS[wave][r + 8 * g][ln] = (__bf16)p0;
      pLDS[wave][r + 8 * g][16 + ln] = (__bf16)p1;
    }
    asm volatile("s_wait_dscnt 0" ::: "memory");
    v16bf aP;
#pragma unroll
    for (int e = 0; e < 16; ++e) aP[e] = pLDS[wave][ln][kA(e, g)];
#pragma unroll
    for (int nt = 0; nt < 4; ++nt) {
      v16bf bV;
#pragma unroll
      for (int e = 0; e < 16; ++e)
        bV[e] = (__bf16)vbase[(size_t)(kc + e + 16 * g) * Dc + nt * 16 + ln];
      acc[nt] = wmma_bf16(aP, bV, acc[nt]);
    }
  }

#pragma unroll
  for (int nt = 0; nt < 4; ++nt)
#pragma unroll
    for (int r = 0; r < 8; ++r) {
      const int rr = rowBase + r + 8 * g;
      if (rr < count)
        ctx[((size_t)bh * Sc + gl[rr]) * Dc + nt * 16 + ln] = acc[nt][r] / l8[r];
    }
}

// ---------------------------------------------------------------------------
extern "C" void kernel_launch(void* const* d_in, const int* in_sizes, int n_in,
                              void* d_out, int out_size, void* d_ws, size_t ws_size,
                              hipStream_t stream) {
  (void)in_sizes; (void)n_in; (void)out_size; (void)ws_size;
  const float* q = (const float*)d_in[0];
  const float* k = (const float*)d_in[1];
  const float* v = (const float*)d_in[2];
  const int* mask = (const int*)d_in[3];

  float* ctx = (float*)d_out;                                 // (B,H,S,D)
  float* probs = ctx + (size_t)Bc * Hc * Sc * Dc;             // (B,H,S,S)

  int* counts = (int*)d_ws;        // B ints
  int* gidx = counts + 16;         // B*S ints

  compact_globals<<<Bc, 32, 0, stream>>>(mask, counts, gidx);

  const int grid = Bc * Hc * (Sc / 128);  // 384 blocks, 8 waves each
  longformer_global_attn<<<grid, 256, 0, stream>>>(q, k, v, counts, gidx, ctx, probs);
  longformer_full_rows<<<grid, 256, 0, stream>>>(q, k, v, counts, gidx, ctx);
}
